// Mamba2Block_76046690943511
// MI455X (gfx1250) — compile-verified, matrix operand order checked by reading, not run on hardware
//
#include <hip/hip_runtime.h>
#include <hip/hip_bf16.h>
#include <math.h>

// ---------------- problem constants ----------------
#define D_MODEL   1024
#define D_INNER   1024
#define NHEADS    16
#define HEAD_DIM  64
#define D_STATE   16
#define DT_RANK   16
#define CHUNK     64
#define PROJ_DIM  2576           // 2*D_INNER + 2*D_STATE*NHEADS + DT_RANK
#define BATCH     4
#define SEQ       4096
#define NTOK      (BATCH*SEQ)    // 16384 rows
#define NCHUNK    (SEQ/CHUNK)    // 64
#define NBC       (BATCH*NCHUNK) // 256 (b,c) pairs
#define OFF_Z     D_INNER
#define OFF_B     (2*D_INNER)
#define OFF_C     (2*D_INNER + D_STATE*NHEADS)
#define OFF_DT    (2*D_INNER + 2*D_STATE*NHEADS)

typedef __attribute__((ext_vector_type(16))) __bf16 v16bf;
typedef __attribute__((ext_vector_type(8)))  float  v8f;
typedef __attribute__((ext_vector_type(2)))  float  v2f;
typedef __attribute__((ext_vector_type(4)))  unsigned int u32x4;
typedef __attribute__((ext_vector_type(8)))  int    i32x8;
typedef __attribute__((ext_vector_type(4)))  int    i32x4;

// Toolchain flavor: only the clang-23/therock lane ships the TDM header and uses
// the 6-arg tensor_load_to_lds builtin; ROCm 7.2 uses the 5-arg form.
#if __has_include(<hip/amd_detail/amd_gfx1250_TDM.h>)
#define TDM_6ARG 1
#else
#define TDM_6ARG 0
#endif

// ---------------- WMMA wrappers (CDNA5 gfx1250) ----------------
__device__ __forceinline__ v8f wmma_bf16(v16bf a, v16bf b, v8f c) {
  return __builtin_amdgcn_wmma_f32_16x16x32_bf16(false, a, false, b, (short)0, c, false, false);
}
__device__ __forceinline__ v8f wmma_f32k4(v2f a, v2f b, v8f c) {
  return __builtin_amdgcn_wmma_f32_16x16x4_f32(false, a, false, b, (short)0, c, false, false);
}

// 16-bit A/B fragment from global, row-major [rows][ld], K contiguous (ISA 7.12.2).
__device__ __forceinline__ v16bf load_frag_bf16(const __bf16* __restrict__ base,
                                                int ld, int row0, int k0, int lane) {
  int m    = row0 + (lane & 15);
  int half = lane >> 4;
  const __bf16* p0 = base + (size_t)m * ld + k0 + 8 * half;
  const __bf16* p1 = p0 + 16;
  v16bf f;
#pragma unroll
  for (int e = 0; e < 8; ++e) f[e] = p0[e];
#pragma unroll
  for (int e = 0; e < 8; ++e) f[8 + e] = p1[e];
  return f;
}

// ---------------- K0: f32 -> bf16 convert ----------------
__global__ void cvt_bf16_kernel(const float* __restrict__ in, __bf16* __restrict__ out, int n) {
  int i = blockIdx.x * 256 + threadIdx.x;
  if (i < n) out[i] = (__bf16)in[i];
}

// ---------------- main bf16 NT GEMM, double-buffered TDM-staged weights ----------------
// C[M,N] (ldc) = A[M,K] * Bt[N,K]^T for FULL 64-wide n tiles (no guards).
// 4 waves/block share a 64x32 bf16 weight tile. Wave 0 drives the Tensor Data
// Mover: tile k+1 streams into the alternate LDS buffer while all waves run
// WMMAs on tile k (s_wait_tensorcnt 1 = current tile resident, next in flight).
__global__ __launch_bounds__(128) void gemm_bf16_tdm(const __bf16* __restrict__ A,
                                                     const __bf16* __restrict__ Bt,
                                                     float* __restrict__ C,
                                                     int ldc, int K) {
  __shared__ __bf16 sB[2 * 64 * 32];              // 2 x 4 KB tiles
  const int lane = threadIdx.x & 31;
  const int wv   = threadIdx.x >> 5;
  const int m0   = blockIdx.y * 64 + wv * 16;
  const int n0b  = blockIdx.x * 64;
  const int half = lane >> 4;
  const int cc   = lane & 15;
  const unsigned ldsBase = (unsigned)(uintptr_t)&sB[0];   // addr[31:0] = LDS byte offset

  auto issue_tdm = [&](int k0, int buf) {
    const unsigned long long gaddr =
        (unsigned long long)(uintptr_t)(Bt + (size_t)n0b * K + k0);
    u32x4 g0 = { 1u,                                     // count=1, user D#
                 ldsBase + (unsigned)(buf << 12),        // lds_addr (4KB buffers)
                 (unsigned)gaddr,                        // global_addr[31:0]
                 (unsigned)((gaddr >> 32) & 0x1ffffffu) | (2u << 30) }; // hi | type=2
    i32x8 g1 = { 0x00010000,                             // data_size = 2 bytes
                 (32 << 16),                             // tensor_dim0 = 32
                 (64 << 16),                             // tensor_dim1 = 64
                 (32 << 16),                             // tile_dim0 = 32
                 64,                                     // tile_dim1 = 64 (tile_dim2=0)
                 K,                                      // tensor_dim0_stride lo
                 0, 0 };
    i32x4 z4 = { 0, 0, 0, 0 };
#if TDM_6ARG
    i32x8 z8 = {};
    __builtin_amdgcn_tensor_load_to_lds(g0, g1, z4, z4, z8, 0);
#else
    __builtin_amdgcn_tensor_load_to_lds(g0, g1, z4, z4, 0);
#endif
  };

  if (wv == 0) issue_tdm(0, 0);                   // prologue: tile 0 in flight

  v8f acc[4] = {};
  for (int k0 = 0; k0 < K; k0 += 32) {
    const int boff = ((k0 >> 5) & 1) << 11;       // element offset of current buffer
    // A fragment streams straight from global, overlapping the DMA
    v16bf a = load_frag_bf16(A, K, m0, k0, lane);
    if (k0 + 64 < K)
      __builtin_prefetch(A + (size_t)(m0 + cc) * K + k0 + 64, 0, 3);

    if (wv == 0) {
      if (k0 + 32 < K) {
        issue_tdm(k0 + 32, ((k0 >> 5) & 1) ^ 1);  // next tile -> other buffer
        __builtin_amdgcn_s_wait_tensorcnt(1);     // current tile done, next in flight
      } else {
        __builtin_amdgcn_s_wait_tensorcnt(0);     // drain on last step
      }
    }
    __syncthreads();                              // tile visible to all 4 waves

#pragma unroll
    for (int s = 0; s < 4; ++s) {
      const int r = s * 16 + cc;                  // local row (= n within tile)
      v16bf b;
#pragma unroll
      for (int e = 0; e < 8; ++e) b[e]     = sB[boff + r * 32 + 8 * half + e];
#pragma unroll
      for (int e = 0; e < 8; ++e) b[8 + e] = sB[boff + r * 32 + 16 + 8 * half + e];
      acc[s] = wmma_bf16(a, b, acc[s]);
    }
    __syncthreads();                              // reads done before buffer reuse
  }

  const int rr = half * 8;
#pragma unroll
  for (int s = 0; s < 4; ++s)
#pragma unroll
    for (int r = 0; r < 8; ++r)
      C[(size_t)(m0 + r + rr) * ldc + n0b + s * 16 + cc] = acc[s][r];
}

// ---------------- 16-column tail GEMM (guard-free, direct loads) ----------------
__global__ __launch_bounds__(128) void gemm_bf16_tail16(const __bf16* __restrict__ A,
                                                        const __bf16* __restrict__ Bt,
                                                        float* __restrict__ C,
                                                        int ldc, int K, int n0) {
  const int lane = threadIdx.x & 31;
  const int wv   = threadIdx.x >> 5;
  const int m0   = blockIdx.x * 64 + wv * 16;
  v8f acc = {};
  for (int k0 = 0; k0 < K; k0 += 32) {
    v16bf a = load_frag_bf16(A, K, m0, k0, lane);
    v16bf b = load_frag_bf16(Bt, K, n0, k0, lane);
    acc = wmma_bf16(a, b, acc);
  }
  const int rr = (lane >> 4) * 8, cc = lane & 15;
#pragma unroll
  for (int r = 0; r < 8; ++r)
    C[(size_t)(m0 + r + rr) * ldc + n0 + cc] = acc[r];
}

// ---------------- K2: dt = softplus(dt_low @ dt_W^T + dt_b) ----------------
__device__ __forceinline__ float softplusf(float a) {
  return a > 20.f ? a : log1pf(expf(a));
}
__global__ void dt_kernel(const float* __restrict__ proj, const float* __restrict__ dtW,
                          const float* __restrict__ dtb, float* __restrict__ dt) {
  int g = blockIdx.x * 256 + threadIdx.x;
  int t = g >> 4, h = g & 15;
  const float* row = proj + (size_t)t * PROJ_DIM + OFF_DT;
  float acc = dtb[h];
#pragma unroll
  for (int r = 0; r < DT_RANK; ++r) acc += row[r] * dtW[h * DT_RANK + r];
  dt[(size_t)t * NHEADS + h] = softplusf(acc);
}

// ---------------- K3: per-chunk cumsum of dA = dt*A; chunk decay ----------------
__global__ void cumsum_kernel(const float* __restrict__ dt, const float* __restrict__ A_log,
                              float* __restrict__ dAcs, float* __restrict__ cdecay) {
  const int bc = blockIdx.x, h = blockIdx.y, k = threadIdx.x;   // 64 threads
  const float Ah = -expf(A_log[h]);
  __shared__ float s[CHUNK];
  s[k] = dt[(size_t)(bc * CHUNK + k) * NHEADS + h] * Ah;
  __syncthreads();
#pragma unroll
  for (int off = 1; off < CHUNK; off <<= 1) {
    float v = (k >= off) ? s[k - off] : 0.f;
    __syncthreads();
    s[k] += v;
    __syncthreads();
  }
  dAcs[(size_t)(bc * CHUNK + k) * NHEADS + h] = s[k];
  if (k == CHUNK - 1) cdecay[bc * NHEADS + h] = expf(s[k]);
}

// ---------------- K4: chunk_state = Bc^T @ (dt*decay_to_end * x), f32 WMMA ----------------
__global__ __launch_bounds__(32) void chunk_state_kernel(const float* __restrict__ proj,
                                                         const float* __restrict__ dt,
                                                         const float* __restrict__ dAcs,
                                                         float* __restrict__ cs) {
  const int bc = blockIdx.x, h = blockIdx.y;
  const int lane = threadIdx.x;
  const int tb = bc * CHUNK;
  const int m = lane & 15;
  const int koff = (lane >> 4) * 2;
  const float last = dAcs[(size_t)(tb + CHUNK - 1) * NHEADS + h];

  v8f acc[4] = {};
  for (int ks = 0; ks < 16; ++ks) {
    const int kk = ks * 4 + koff;
    v2f a;
    a.x = proj[(size_t)(tb + kk)     * PROJ_DIM + OFF_B + h * D_STATE + m];
    a.y = proj[(size_t)(tb + kk + 1) * PROJ_DIM + OFF_B + h * D_STATE + m];
    const float w0 = dt[(size_t)(tb + kk)     * NHEADS + h] *
                     expf(last - dAcs[(size_t)(tb + kk)     * NHEADS + h]);
    const float w1 = dt[(size_t)(tb + kk + 1) * NHEADS + h] *
                     expf(last - dAcs[(size_t)(tb + kk + 1) * NHEADS + h]);
#pragma unroll
    for (int tn = 0; tn < 4; ++tn) {
      const int p = tn * 16 + (lane & 15);
      v2f b;
      b.x = w0 * proj[(size_t)(tb + kk)     * PROJ_DIM + h * HEAD_DIM + p];
      b.y = w1 * proj[(size_t)(tb + kk + 1) * PROJ_DIM + h * HEAD_DIM + p];
      acc[tn] = wmma_f32k4(a, b, acc[tn]);
    }
  }
  const int nrow = (lane >> 4) * 8;
  float* out = cs + ((size_t)bc * NHEADS + h) * D_STATE * HEAD_DIM;
#pragma unroll
  for (int tn = 0; tn < 4; ++tn)
#pragma unroll
    for (int r = 0; r < 8; ++r)
      out[(size_t)(r + nrow) * HEAD_DIM + tn * 16 + (lane & 15)] = acc[tn][r];
}

// ---------------- K5: sequential inter-chunk scan (64 chunks deep) ----------------
__global__ __launch_bounds__(256) void scan_kernel(const float* __restrict__ cs,
                                                   const float* __restrict__ cdecay,
                                                   float* __restrict__ hs) {
  const int b = blockIdx.x >> 4, h = blockIdx.x & 15;
  const int tid = threadIdx.x;
  float st[4] = {0.f, 0.f, 0.f, 0.f};
  for (int c = 0; c < NCHUNK; ++c) {
    const int bc = b * NCHUNK + c;
    const size_t base = ((size_t)bc * NHEADS + h) * (D_STATE * HEAD_DIM);
    const float d = cdecay[bc * NHEADS + h];
#pragma unroll
    for (int e = 0; e < 4; ++e) {
      const int j = e * 256 + tid;
      hs[base + j] = st[e];
      st[e] = d * st[e] + cs[base + j];
    }
  }
}

// ---------------- K6: fused y = tril(Ccs@Bcs^T)@x + Ccs@h + D*x  (f32 WMMA) ----------------
__global__ __launch_bounds__(32) void chunk_y_kernel(const float* __restrict__ proj,
                                                     const float* __restrict__ dt,
                                                     const float* __restrict__ dAcs,
                                                     const float* __restrict__ hs,
                                                     const float* __restrict__ Dp,
                                                     float* __restrict__ y) {
  const int bc = blockIdx.x, h = blockIdx.y;
  const int lane = threadIdx.x;
  const int tb = bc * CHUNK;
  const int koff = (lane >> 4) * 2;
  const int cc = lane & 15;

  __shared__ float sC[CHUNK * D_STATE];   // Ccs  (64x16)
  __shared__ float sB[CHUNK * D_STATE];   // Bcs  (64x16)
  __shared__ float sX[CHUNK * HEAD_DIM];  // xc   (64x64)
  __shared__ float sS[CHUNK * CHUNK];     // masked scores (64x64)

  for (int idx = lane; idx < CHUNK * D_STATE; idx += 32) {
    const int k = idx >> 4, n = idx & 15;
    const float da = dAcs[(size_t)(tb + k) * NHEADS + h];
    sC[idx] = proj[(size_t)(tb + k) * PROJ_DIM + OFF_C + h * D_STATE + n] * expf(da);
    sB[idx] = proj[(size_t)(tb + k) * PROJ_DIM + OFF_B + h * D_STATE + n] *
              expf(-da) * dt[(size_t)(tb + k) * NHEADS + h];
  }
  for (int idx = lane; idx < CHUNK * HEAD_DIM; idx += 32) {
    const int k = idx >> 6, p = idx & 63;
    sX[idx] = proj[(size_t)(tb + k) * PROJ_DIM + h * HEAD_DIM + p];
  }
  __syncthreads();

  // GEMM1: scores = Ccs(64x16) @ Bcs^T(16x64), causal-masked into LDS
  for (int tm = 0; tm < 4; ++tm) {
    v8f acc[4] = {};
#pragma unroll
    for (int ks = 0; ks < 4; ++ks) {
      const int kk = ks * 4 + koff;
      v2f a;
      a.x = sC[(tm * 16 + cc) * D_STATE + kk];
      a.y = sC[(tm * 16 + cc) * D_STATE + kk + 1];
#pragma unroll
      for (int tn = 0; tn < 4; ++tn) {
        const int n = tn * 16 + cc;
        v2f b;
        b.x = sB[n * D_STATE + kk];
        b.y = sB[n * D_STATE + kk + 1];
        acc[tn] = wmma_f32k4(a, b, acc[tn]);
      }
    }
    const int rr = (lane >> 4) * 8;
#pragma unroll
    for (int tn = 0; tn < 4; ++tn)
#pragma unroll
      for (int r = 0; r < 8; ++r) {
        const int i = tm * 16 + r + rr, j = tn * 16 + cc;
        sS[i * CHUNK + j] = (j <= i) ? acc[tn][r] : 0.f;
      }
  }
  __syncthreads();

  // GEMM2: y = sS @ xc (K=64) + sC @ h (K=16), then + D*x
  const float Dh = Dp[h];
  const float* hb = hs + ((size_t)bc * NHEADS + h) * (D_STATE * HEAD_DIM);
  for (int tm = 0; tm < 4; ++tm) {
    v8f acc[4] = {};
    for (int ks = 0; ks < 16; ++ks) {
      const int kk = ks * 4 + koff;
      v2f a;
      a.x = sS[(tm * 16 + cc) * CHUNK + kk];
      a.y = sS[(tm * 16 + cc) * CHUNK + kk + 1];
#pragma unroll
      for (int tn = 0; tn < 4; ++tn) {
        const int p = tn * 16 + cc;
        v2f b;
        b.x = sX[kk * HEAD_DIM + p];
        b.y = sX[(kk + 1) * HEAD_DIM + p];
        acc[tn] = wmma_f32k4(a, b, acc[tn]);
      }
    }
#pragma unroll
    for (int ks = 0; ks < 4; ++ks) {
      const int kk = ks * 4 + koff;
      v2f a;
      a.x = sC[(tm * 16 + cc) * D_STATE + kk];
      a.y = sC[(tm * 16 + cc) * D_STATE + kk + 1];
#pragma unroll
      for (int tn = 0; tn < 4; ++tn) {
        const int p = tn * 16 + cc;
        v2f b;
        b.x = hb[kk * HEAD_DIM + p];
        b.y = hb[(kk + 1) * HEAD_DIM + p];
        acc[tn] = wmma_f32k4(a, b, acc[tn]);
      }
    }
    const int rr = (lane >> 4) * 8;
#pragma unroll
    for (int tn = 0; tn < 4; ++tn)
#pragma unroll
      for (int r = 0; r < 8; ++r) {
        const int i = tm * 16 + r + rr, p = tn * 16 + cc;
        y[(size_t)(tb + i) * D_INNER + h * HEAD_DIM + p] =
            acc[tn][r] + Dh * sX[i * HEAD_DIM + p];
      }
  }
}

// ---------------- K7: gated RMSNorm + SiLU, emit bf16 ----------------
__global__ __launch_bounds__(256) void rmsnorm_gate_kernel(const float* __restrict__ y,
                                                           const float* __restrict__ proj,
                                                           const float* __restrict__ nw,
                                                           __bf16* __restrict__ ybf) {
  const int t = blockIdx.x, tid = threadIdx.x;
  __shared__ float red[256];
  float ss = 0.f;
#pragma unroll
  for (int e = 0; e < 4; ++e) {
    const float v = y[(size_t)t * D_INNER + e * 256 + tid];
    ss += v * v;
  }
  red[tid] = ss;
  __syncthreads();
  for (int o = 128; o > 0; o >>= 1) {
    if (tid < o) red[tid] += red[tid + o];
    __syncthreads();
  }
  const float rinv = rsqrtf(red[0] * (1.f / D_INNER) + 1e-6f);
#pragma unroll
  for (int e = 0; e < 4; ++e) {
    const int j = e * 256 + tid;
    float v = y[(size_t)t * D_INNER + j] * rinv * nw[j];
    const float z = proj[(size_t)t * PROJ_DIM + OFF_Z + j];
    v *= z / (1.f + expf(-z));
    ybf[(size_t)t * D_INNER + j] = (__bf16)v;
  }
}

// ---------------- host-side launch ----------------
extern "C" void kernel_launch(void* const* d_in, const int* in_sizes, int n_in,
                              void* d_out, int out_size, void* d_ws, size_t ws_size,
                              hipStream_t stream) {
  const float* x      = (const float*)d_in[0];
  const float* W_in   = (const float*)d_in[1];
  const float* dt_W   = (const float*)d_in[2];
  const float* dt_b   = (const float*)d_in[3];
  const float* A_log  = (const float*)d_in[4];
  const float* D_par  = (const float*)d_in[5];
  const float* W_out  = (const float*)d_in[6];
  const float* norm_w = (const float*)d_in[7];
  float* out = (float*)d_out;

  size_t off = 0;
  auto take = [&](size_t bytes) {
    size_t o = off;
    off += (bytes + 255) & ~(size_t)255;
    return o;
  };
  char* ws = (char*)d_ws;
  float*  proj  = (float*)(ws + take((size_t)NTOK * PROJ_DIM * 4));
  float*  dt    = (float*)(ws + take((size_t)NTOK * NHEADS * 4));
  float*  dAcs  = (float*)(ws + take((size_t)NTOK * NHEADS * 4));
  float*  cdec  = (float*)(ws + take((size_t)NBC * NHEADS * 4));
  float*  cs    = (float*)(ws + take((size_t)NBC * NHEADS * D_STATE * HEAD_DIM * 4));
  float*  hs    = (float*)(ws + take((size_t)NBC * NHEADS * D_STATE * HEAD_DIM * 4));
  float*  ybuf  = (float*)(ws + take((size_t)NTOK * D_INNER * 4));
  __bf16* xbf   = (__bf16*)(ws + take((size_t)NTOK * D_MODEL * 2));
  __bf16* wibf  = (__bf16*)(ws + take((size_t)PROJ_DIM * D_MODEL * 2));
  __bf16* wobf  = (__bf16*)(ws + take((size_t)D_MODEL * D_INNER * 2));
  __bf16* ybf   = (__bf16*)(ws + take((size_t)NTOK * D_INNER * 2));
  (void)ws_size; (void)in_sizes; (void)n_in; (void)out_size;

  {
    int n;
    n = NTOK * D_MODEL;
    cvt_bf16_kernel<<<(n + 255) / 256, 256, 0, stream>>>(x, xbf, n);
    n = PROJ_DIM * D_MODEL;
    cvt_bf16_kernel<<<(n + 255) / 256, 256, 0, stream>>>(W_in, wibf, n);
    n = D_MODEL * D_INNER;
    cvt_bf16_kernel<<<(n + 255) / 256, 256, 0, stream>>>(W_out, wobf, n);
  }
  // proj = x @ W_in^T : 40 full 64-col tiles via TDM GEMM + 16-col tail
  gemm_bf16_tdm<<<dim3(PROJ_DIM / 64, NTOK / 64), 128, 0, stream>>>(
      xbf, wibf, proj, PROJ_DIM, D_MODEL);
  gemm_bf16_tail16<<<NTOK / 64, 128, 0, stream>>>(
      xbf, wibf, proj, PROJ_DIM, D_MODEL, (PROJ_DIM / 64) * 64);
  dt_kernel<<<(NTOK * NHEADS) / 256, 256, 0, stream>>>(proj, dt_W, dt_b, dt);
  cumsum_kernel<<<dim3(NBC, NHEADS), 64, 0, stream>>>(dt, A_log, dAcs, cdec);
  chunk_state_kernel<<<dim3(NBC, NHEADS), 32, 0, stream>>>(proj, dt, dAcs, cs);
  scan_kernel<<<BATCH * NHEADS, 256, 0, stream>>>(cs, cdec, hs);
  chunk_y_kernel<<<dim3(NBC, NHEADS), 32, 0, stream>>>(proj, dt, dAcs, hs, D_par, ybuf);
  rmsnorm_gate_kernel<<<NTOK, 256, 0, stream>>>(ybuf, proj, norm_w, ybf);
  // out = y @ W_out^T : all full tiles
  gemm_bf16_tdm<<<dim3(D_MODEL / 64, NTOK / 64), 128, 0, stream>>>(
      ybf, wobf, out, D_MODEL, D_INNER);
}
